// Masked_MHA_74010876444743
// MI455X (gfx1250) — matrix operand order repaired and checked by reading, not compiled, not run on hardware
//
#include <hip/hip_runtime.h>
#include <hip/hip_bf16.h>

typedef __attribute__((ext_vector_type(16))) _Float16 v16h;
typedef __attribute__((ext_vector_type(8)))  _Float16 v8h;
typedef __attribute__((ext_vector_type(8)))  float    v8f;
typedef __attribute__((ext_vector_type(4)))  unsigned int v4u;
typedef __attribute__((ext_vector_type(8)))  int      v8i;
typedef __attribute__((ext_vector_type(4)))  int      v4i;

#define EMBED 1024
#define HEADS 16
#define DK    64
#define SEQL  2048
#define BATCH 2
#define ROWS  (BATCH * SEQL)          // 4096 token rows

#if defined(__has_builtin)
#if __has_builtin(__builtin_amdgcn_tensor_load_to_lds)
#define HAVE_TDM 1
#endif
#endif
#ifndef HAVE_TDM
#define HAVE_TDM 0
#endif

// ---------------------------------------------------------------------------
// WMMA helpers (gfx1250 wave32, 16x16x32 f16 -> f32 accumulate)
// ---------------------------------------------------------------------------
__device__ __forceinline__ v8f wmma_f16(v16h a, v16h b, v8f c) {
  return __builtin_amdgcn_wmma_f32_16x16x32_f16(false, a, false, b, (short)0, c,
                                                false, false);
}

// A fragment: 16x32 (MxK) tile of a row-major f16 matrix, leading dim lda.
// lane -> M = lane&15; K = 16*(lane>>4) + 0..15 contiguous.
// Two b128 loads per lane.
__device__ __forceinline__ v16h load_a_frag(const _Float16* base, int lda) {
  unsigned lane = threadIdx.x & 31u;
  unsigned koff = (lane >> 4) << 4;                       // 0 or 16
  const _Float16* row = base + (size_t)(lane & 15u) * (size_t)lda;
  v8h lo = *reinterpret_cast<const v8h*>(row + koff);
  v8h hi = *reinterpret_cast<const v8h*>(row + koff + 8);
  return __builtin_shufflevector(lo, hi, 0, 1, 2, 3, 4, 5, 6, 7,
                                 8, 9, 10, 11, 12, 13, 14, 15);
}

// B fragment: 32x16 (KxN) with B(k,n) = M[n][k] of a row-major matrix M.
// lane -> N = lane&15; K = 16*(lane>>4) + 0..15 contiguous -> two b128 loads.
__device__ __forceinline__ v16h load_bT_frag(const _Float16* base, int ld) {
  unsigned lane = threadIdx.x & 31u;
  const _Float16* row =
      base + (size_t)(lane & 15u) * (size_t)ld + ((lane >> 4) << 4);
  v8h lo = *reinterpret_cast<const v8h*>(row);
  v8h hi = *reinterpret_cast<const v8h*>(row + 8);
  return __builtin_shufflevector(lo, hi, 0, 1, 2, 3, 4, 5, 6, 7,
                                 8, 9, 10, 11, 12, 13, 14, 15);
}

// ---------------------------------------------------------------------------
// Tensor Data Mover: 2-D f16 tile DMA global -> LDS (D# per ISA 8.3-8.6).
// tile_d0 = contiguous elements per row, tile_d1 = rows,
// stride = row stride in elements.  Tiles are always fully in-bounds, so
// tensor_dim == tile_dim.  Groups 2/3 zero (<=2-D tensor).
// ---------------------------------------------------------------------------
__device__ __forceinline__ void tdm_load_tile_f16(unsigned lds_byte_addr,
                                                  const _Float16* gsrc,
                                                  unsigned tile_d0,
                                                  unsigned tile_d1,
                                                  unsigned stride_elems) {
#if HAVE_TDM
  unsigned long long ga = (unsigned long long)(uintptr_t)gsrc;
  v4u g0;
  g0[0] = 1u;                                  // count=1 (valid), user mode
  g0[1] = lds_byte_addr;                       // lds_addr
  g0[2] = (unsigned)ga;                        // global_addr[31:0]
  g0[3] = (unsigned)((ga >> 32) & 0x01ffffffu) | (2u << 30);  // addr hi, type=2
  v8i g1;
  g1[0] = (int)(1u << 16);                     // data_size=1 -> 2 bytes
  g1[1] = (int)(tile_d0 << 16);                // tensor_dim0[15:0]
  g1[2] = (int)(tile_d1 << 16);                // tensor_dim1[15:0]
  g1[3] = (int)(tile_d0 << 16);                // tile_dim0
  g1[4] = (int)tile_d1;                        // tile_dim1 (tile_dim2=0)
  g1[5] = (int)stride_elems;                   // tensor_dim0_stride[31:0]
  g1[6] = 0;                                   // stride hi / dim1_stride lo
  g1[7] = 0;
  v4i g2 = {0, 0, 0, 0};
  v4i g3 = {0, 0, 0, 0};
#if __clang_major__ >= 23
  v8i g4 = {0, 0, 0, 0, 0, 0, 0, 0};
  __builtin_amdgcn_tensor_load_to_lds(g0, g1, g2, g3, g4, 0);
#else
  __builtin_amdgcn_tensor_load_to_lds(g0, g1, g2, g3, 0);
#endif
#else
  (void)lds_byte_addr; (void)gsrc; (void)tile_d0; (void)tile_d1;
  (void)stride_elems;
#endif
}

__device__ __forceinline__ void wait_tensorcnt_le2() {
#if __has_builtin(__builtin_amdgcn_s_wait_tensorcnt)
  __builtin_amdgcn_s_wait_tensorcnt(2);
#else
  asm volatile("s_wait_tensorcnt 0x2" ::: "memory");
#endif
}
__device__ __forceinline__ void wait_tensorcnt_zero() {
#if __has_builtin(__builtin_amdgcn_s_wait_tensorcnt)
  __builtin_amdgcn_s_wait_tensorcnt(0);
#else
  asm volatile("s_wait_tensorcnt 0x0" ::: "memory");
#endif
}

// ---------------------------------------------------------------------------
// f32 -> f16 conversion (grid-stride)
// ---------------------------------------------------------------------------
__global__ void f32_to_f16_kernel(const float* __restrict__ in,
                                  _Float16* __restrict__ out, size_t n) {
  size_t i = (size_t)blockIdx.x * blockDim.x + threadIdx.x;
  size_t stride = (size_t)gridDim.x * blockDim.x;
  for (; i < n; i += stride) out[i] = (_Float16)in[i];
}

// ---------------------------------------------------------------------------
// QKV projection + bias + RoPE.  One wave = 16 token rows x one full head.
// grid = (ROWS/16/4, HEADS, 3[q,k,v]), block = 128 (4 waves)
// ---------------------------------------------------------------------------
__global__ __launch_bounds__(128) void qkv_rope_kernel(
    const _Float16* __restrict__ Xh,   // [ROWS][EMBED]
    const _Float16* __restrict__ Wqh,  // [EMBED][EMBED] row-major (out,in)
    const _Float16* __restrict__ Wkh,
    const _Float16* __restrict__ Wvh,
    const float* __restrict__ bq, const float* __restrict__ bk,
    const float* __restrict__ bv,
    const float* __restrict__ cosp,    // [SEQL][DK]
    const float* __restrict__ sinp,
    _Float16* __restrict__ Qr,         // [B*H][SEQL][DK]
    _Float16* __restrict__ Kr,         // [B*H][SEQL][DK]
    _Float16* __restrict__ Vt)         // [B*H][DK][SEQL]  (transposed)
{
  const int wave = threadIdx.x >> 5;
  const unsigned lane = threadIdx.x & 31u;
  const int rowTile = blockIdx.x * 4 + wave;
  const int h = blockIdx.y;
  const int which = blockIdx.z;
  const int row0 = rowTile * 16;

  const _Float16* W = (which == 0) ? Wqh : (which == 1) ? Wkh : Wvh;
  const float* bias = (which == 0) ? bq : (which == 1) ? bk : bv;

  v8f acc[4] = {};
  for (int k = 0; k < EMBED; k += 32) {
    v16h a = load_a_frag(Xh + (size_t)row0 * EMBED + k, EMBED);
#pragma unroll
    for (int t = 0; t < 4; ++t) {
      v16h b = load_bT_frag(W + (size_t)(h * DK + t * 16) * EMBED + k, EMBED);
      acc[t] = wmma_f16(a, b, acc[t]);
    }
  }

  const int half8 = (lane >> 4) << 3;
  const int cl = lane & 15;
  const int b = row0 / SEQL;
  const int s_base = row0 % SEQL;

  if (which < 2) {
    float res[4][8];
#pragma unroll
    for (int t = 0; t < 4; ++t) {
      int d = t * 16 + cl;
      float bcur = bias[h * DK + d];
      float bpar = bias[h * DK + ((t < 2) ? d + 32 : d - 32)];
#pragma unroll
      for (int r = 0; r < 8; ++r) {
        int s = s_base + half8 + r;
        float c = cosp[(size_t)s * DK + d];
        float sn = sinp[(size_t)s * DK + d];
        float x = acc[t][r] + bcur;
        float xr = (t < 2) ? -(acc[t + 2][r] + bpar) : (acc[t - 2][r] + bpar);
        res[t][r] = x * c + xr * sn;
      }
    }
    _Float16* dst =
        ((which == 0) ? Qr : Kr) + (size_t)(b * HEADS + h) * SEQL * DK;
#pragma unroll
    for (int t = 0; t < 4; ++t)
#pragma unroll
      for (int r = 0; r < 8; ++r)
        dst[(size_t)(s_base + half8 + r) * DK + t * 16 + cl] =
            (_Float16)res[t][r];
  } else {
    _Float16* dst = Vt + (size_t)(b * HEADS + h) * DK * SEQL;
#pragma unroll
    for (int t = 0; t < 4; ++t) {
      int d = t * 16 + cl;
      float bcur = bias[h * DK + d];
#pragma unroll
      for (int r = 0; r < 8; ++r)
        dst[(size_t)d * SEQL + (s_base + half8 + r)] =
            (_Float16)(acc[t][r] + bcur);
    }
  }
}

// ---------------------------------------------------------------------------
// Flash causal attention with TDM-staged K/V tiles.
// Block = 4 waves = 64 query rows of one (b,h).  Per 32-key block, wave 0
// issues two TDM tile loads (K: 32x64 contiguous-row tile; V^T: 64x32 tile
// with row stride SEQL) into a double buffer, waits TENSORcnt, barrier;
// all 4 waves then run 4 score WMMAs + online softmax + 4 P*V WMMAs from LDS.
// grid = (B*H*SEQL/64), block = 128
// ---------------------------------------------------------------------------
__global__ __launch_bounds__(128) void attn_kernel(
    const _Float16* __restrict__ Q,   // [B*H][SEQL][DK]
    const _Float16* __restrict__ K,   // [B*H][SEQL][DK]
    const _Float16* __restrict__ Vt,  // [B*H][DK][SEQL]
    _Float16* __restrict__ Oh)        // [B][SEQL][H][DK]
{
  __shared__ __align__(16) _Float16 Kbuf[2][32 * DK];   // 2 x 4 KiB
  __shared__ __align__(16) _Float16 Vbuf[2][DK * 32];   // 2 x 4 KiB
  __shared__ __align__(16) _Float16 Pt[4][16 * 32];     // 4 x 1 KiB

  const int wave = threadIdx.x >> 5;
  const unsigned lane = threadIdx.x & 31u;
  const int bh = blockIdx.x >> 5;            // /(SEQL/64)
  const int qblk = blockIdx.x & 31;
  const int i0_blk = qblk * 64;
  const int iw0 = i0_blk + wave * 16;        // this wave's query tile

  const _Float16* Qb = Q + (size_t)bh * SEQL * DK;
  const _Float16* Kb = K + (size_t)bh * SEQL * DK;
  const _Float16* Vb = Vt + (size_t)bh * DK * SEQL;

  v16h qa0 = load_a_frag(Qb + (size_t)iw0 * DK, DK);
  v16h qa1 = load_a_frag(Qb + (size_t)iw0 * DK + 32, DK);

  v8f acc[4] = {};
  float m[8], l[8];
#pragma unroll
  for (int r = 0; r < 8; ++r) { m[r] = -1e30f; l[r] = 0.0f; }

  const int half8 = (lane >> 4) << 3;
  const int cl = lane & 15;
  _Float16* Pw = &Pt[wave][0];
  const float scale = 0.125f;                // 1/sqrt(64)

  const int njb = qblk * 2 + 2;              // # of 32-key blocks for block

#if HAVE_TDM
  if (wave == 0) {
    tdm_load_tile_f16((unsigned)(uintptr_t)&Kbuf[0][0], Kb, DK, 32, DK);
    tdm_load_tile_f16((unsigned)(uintptr_t)&Vbuf[0][0], Vb, 32, DK, SEQL);
  }
#endif

  for (int jb = 0; jb < njb; ++jb) {
    const int j = jb * 32;
#if HAVE_TDM
    const int cur = jb & 1;
    if (wave == 0) {
      if (jb + 1 < njb) {
        const int jn = j + 32;
        tdm_load_tile_f16((unsigned)(uintptr_t)&Kbuf[1 - cur][0],
                          Kb + (size_t)jn * DK, DK, 32, DK);
        tdm_load_tile_f16((unsigned)(uintptr_t)&Vbuf[1 - cur][0],
                          Vb + jn, 32, DK, SEQL);
        wait_tensorcnt_le2();                // oldest pair (this jb) done
      } else {
        wait_tensorcnt_zero();
      }
    }
    __syncthreads();                         // LDS tiles ready for all waves
#else
    const int cur = 0;
    __syncthreads();                         // previous compute done
    {
      // cooperative synchronous staging fallback
      for (int i = threadIdx.x * 8; i < 32 * DK; i += 128 * 8)
        *(v8h*)&Kbuf[0][i] = *(const v8h*)(Kb + (size_t)j * DK + i);
      int vrow = threadIdx.x >> 2;           // 0..31 (two passes for 64 rows)
      int vseg = (threadIdx.x & 3) * 8;
      *(v8h*)&Vbuf[0][vrow * 32 + vseg] =
          *(const v8h*)(Vb + (size_t)vrow * SEQL + j + vseg);
      *(v8h*)&Vbuf[0][(vrow + 32) * 32 + vseg] =
          *(const v8h*)(Vb + (size_t)(vrow + 32) * SEQL + j + vseg);
    }
    __syncthreads();
#endif

    if (j <= iw0 + 15) {                     // causal: block has live keys
      const _Float16* Kl = &Kbuf[cur][0];
      const _Float16* Vl = &Vbuf[cur][0];
      v8f s0 = {}, s1 = {};
      s0 = wmma_f16(qa0, load_bT_frag(Kl, DK), s0);
      s0 = wmma_f16(qa1, load_bT_frag(Kl + 32, DK), s0);
      s1 = wmma_f16(qa0, load_bT_frag(Kl + 16 * DK, DK), s1);
      s1 = wmma_f16(qa1, load_bT_frag(Kl + 16 * DK + 32, DK), s1);

      const int col0 = j + cl, col1 = col0 + 16;
#pragma unroll
      for (int r = 0; r < 8; ++r) {
        const int row = iw0 + half8 + r;
        float x0 = (col0 <= row) ? s0[r] * scale : -1e30f;
        float x1 = (col1 <= row) ? s1[r] * scale : -1e30f;
        float mx = fmaxf(x0, x1);
#pragma unroll
        for (int off = 1; off < 16; off <<= 1)
          mx = fmaxf(mx, __shfl_xor(mx, off, 32));
        float mnew = fmaxf(m[r], mx);
        float p0 = __expf(x0 - mnew);
        float p1 = __expf(x1 - mnew);
        float rs = p0 + p1;
#pragma unroll
        for (int off = 1; off < 16; off <<= 1) rs += __shfl_xor(rs, off, 32);
        float alpha = __expf(m[r] - mnew);
        m[r] = mnew;
        l[r] = l[r] * alpha + rs;
        acc[0][r] *= alpha; acc[1][r] *= alpha;
        acc[2][r] *= alpha; acc[3][r] *= alpha;
        Pw[(half8 + r) * 32 + cl] = (_Float16)p0;
        Pw[(half8 + r) * 32 + cl + 16] = (_Float16)p1;
      }

      v16h pa = load_a_frag(Pw, 32);
#pragma unroll
      for (int t = 0; t < 4; ++t) {
        v16h vb = load_bT_frag(Vl + (t * 16) * 32, 32);
        acc[t] = wmma_f16(pa, vb, acc[t]);
      }
    }
    __syncthreads();                         // all reads of buf[cur] done
  }

  const int b = bh >> 4, h = bh & 15;
#pragma unroll
  for (int t = 0; t < 4; ++t)
#pragma unroll
    for (int r = 0; r < 8; ++r) {
      float val = acc[t][r] / l[r];
      int s = iw0 + half8 + r;
      Oh[(((size_t)b * SEQL + s) * HEADS + h) * DK + t * 16 + cl] =
          (_Float16)val;
    }
}

// ---------------------------------------------------------------------------
// Output projection: out = Ah @ Wo^T + bo, f32 result.
// grid = (ROWS/16/4, EMBED/64), block = 128
// ---------------------------------------------------------------------------
__global__ __launch_bounds__(128) void oproj_kernel(
    const _Float16* __restrict__ Ah,   // [ROWS][EMBED] f16
    const _Float16* __restrict__ Woh,  // [EMBED][EMBED] f16 row-major
    const float* __restrict__ bo, float* __restrict__ out)
{
  const int wave = threadIdx.x >> 5;
  const unsigned lane = threadIdx.x & 31u;
  const int rowTile = blockIdx.x * 4 + wave;
  const int n0 = blockIdx.y * 64;
  const int row0 = rowTile * 16;

  v8f acc[4] = {};
  for (int k = 0; k < EMBED; k += 32) {
    v16h a = load_a_frag(Ah + (size_t)row0 * EMBED + k, EMBED);
#pragma unroll
    for (int t = 0; t < 4; ++t) {
      v16h b = load_bT_frag(Woh + (size_t)(n0 + t * 16) * EMBED + k, EMBED);
      acc[t] = wmma_f16(a, b, acc[t]);
    }
  }

  const int half8 = (lane >> 4) << 3;
  const int cl = lane & 15;
#pragma unroll
  for (int t = 0; t < 4; ++t) {
    int n = n0 + t * 16 + cl;
    float bn = bo[n];
#pragma unroll
    for (int r = 0; r < 8; ++r)
      out[(size_t)(row0 + half8 + r) * EMBED + n] = acc[t][r] + bn;
  }
}

// ---------------------------------------------------------------------------
// Host-side launch
// ---------------------------------------------------------------------------
extern "C" void kernel_launch(void* const* d_in, const int* in_sizes, int n_in,
                              void* d_out, int out_size, void* d_ws,
                              size_t ws_size, hipStream_t stream) {
  (void)in_sizes; (void)n_in; (void)out_size; (void)ws_size;

  const float* x    = (const float*)d_in[0];
  /* d_in[1] = mask: causal, handled analytically */
  const float* cosp = (const float*)d_in[2];
  const float* sinp = (const float*)d_in[3];
  const float* Wq   = (const float*)d_in[4];
  const float* bq   = (const float*)d_in[5];
  const float* Wk   = (const float*)d_in[6];
  const float* bk   = (const float*)d_in[7];
  const float* Wv   = (const float*)d_in[8];
  const float* bv   = (const float*)d_in[9];
  const float* Wo   = (const float*)d_in[10];
  const float* bo   = (const float*)d_in[11];

  constexpr size_t SZ_X = (size_t)ROWS * EMBED * 2;               // 8 MiB
  constexpr size_t SZ_W = (size_t)EMBED * EMBED * 2;              // 2 MiB
  constexpr size_t SZ_T = (size_t)BATCH * HEADS * SEQL * DK * 2;  // 8 MiB
  char* ws = (char*)d_ws;
  _Float16* Xh  = (_Float16*)(ws);
  _Float16* Wqh = (_Float16*)(ws + SZ_X);
  _Float16* Wkh = (_Float16*)(ws + SZ_X + SZ_W);
  _Float16* Wvh = (_Float16*)(ws + SZ_X + 2 * SZ_W);
  _Float16* Woh = (_Float16*)(ws + SZ_X + 3 * SZ_W);
  _Float16* Qr  = (_Float16*)(ws + SZ_X + 4 * SZ_W);
  _Float16* Kr  = (_Float16*)(ws + SZ_X + 4 * SZ_W + SZ_T);
  _Float16* Vtw = (_Float16*)(ws + SZ_X + 4 * SZ_W + 2 * SZ_T);
  _Float16* Oh  = (_Float16*)(ws + SZ_X + 4 * SZ_W + 3 * SZ_T);

  f32_to_f16_kernel<<<2048, 256, 0, stream>>>(x, Xh, (size_t)ROWS * EMBED);
  f32_to_f16_kernel<<<1024, 256, 0, stream>>>(Wq, Wqh, (size_t)EMBED * EMBED);
  f32_to_f16_kernel<<<1024, 256, 0, stream>>>(Wk, Wkh, (size_t)EMBED * EMBED);
  f32_to_f16_kernel<<<1024, 256, 0, stream>>>(Wv, Wvh, (size_t)EMBED * EMBED);
  f32_to_f16_kernel<<<1024, 256, 0, stream>>>(Wo, Woh, (size_t)EMBED * EMBED);

  qkv_rope_kernel<<<dim3(ROWS / 16 / 4, HEADS, 3), 128, 0, stream>>>(
      Xh, Wqh, Wkh, Wvh, bq, bk, bv, cosp, sinp, Qr, Kr, Vtw);

  attn_kernel<<<dim3(BATCH * HEADS * (SEQL / 64)), 128, 0, stream>>>(
      Qr, Kr, Vtw, Oh);

  oproj_kernel<<<dim3(ROWS / 16 / 4, EMBED / 64), 128, 0, stream>>>(
      Oh, Woh, bo, (float*)d_out);
}
